// SLULatticeRNN_31121333027061
// MI455X (gfx1250) — matrix operand order repaired and checked by reading, not compiled, not run on hardware
//
#include <hip/hip_runtime.h>
#include <hip/hip_bf16.h>
#include <math.h>

#define BDIM 32     // batch
#define SLEN 256    // sequence
#define KLAT 3      // lattice edges
#define DDIM 512    // embed dim
#define HDIM 512    // hidden
#define GDIM 2048   // 4*H gate dim
#define LOUT 128    // output classes
#define NWG  32     // persistent workgroups in recurrent kernel (16 h-cols each)

typedef __attribute__((ext_vector_type(16))) __bf16 v16bf;
typedef __attribute__((ext_vector_type(8)))  __bf16 v8bf;
typedef __attribute__((ext_vector_type(8)))  float  v8f;

union Frag { v16bf v; v8bf h[2]; unsigned u[8]; };

// Hardware packed f32->bf16 conversion: dst = {bf16(lo), bf16(hi)} in one VALU op.
__device__ __forceinline__ unsigned cvt2(float lo, float hi) {
  unsigned r;
  asm("v_cvt_pk_bf16_f32 %0, %1, %2" : "=v"(r) : "v"(lo), "v"(hi));
  return r;
}

__device__ __forceinline__ float sigm(float x) { return 1.0f / (1.0f + __expf(-x)); }

__device__ __forceinline__ v8f wmma_bf16(v16bf a, v16bf b, v8f c) {
  return __builtin_amdgcn_wmma_f32_16x16x32_bf16(false, a, false, b, (short)0, c, false, false);
}

// ---------------------------------------------------------------------------
// Kernel 1: convert Wih/Whh to bf16 (packed), bias = bih+bhh, zero barrier.
// ---------------------------------------------------------------------------
__global__ __launch_bounds__(256) void k_prep(const float* __restrict__ Wih,
                                              const float* __restrict__ Whh,
                                              const float* __restrict__ bih,
                                              const float* __restrict__ bhh,
                                              unsigned* __restrict__ WihB2,
                                              unsigned* __restrict__ WhhB2,
                                              float* __restrict__ bias,
                                              unsigned* __restrict__ syncp) {
  int tid = blockIdx.x * blockDim.x + threadIdx.x;
  int nthr = gridDim.x * blockDim.x;
  if (tid == 0) { syncp[0] = 0u; syncp[1] = 0u; }
  const int npair = GDIM * DDIM / 2;
  for (int i = tid; i < npair; i += nthr) {
    WihB2[i] = cvt2(Wih[2 * i], Wih[2 * i + 1]);
    WhhB2[i] = cvt2(Whh[2 * i], Whh[2 * i + 1]);
  }
  for (int i = tid; i < GDIM; i += nthr) bias[i] = bih[i] + bhh[i];
}

// ---------------------------------------------------------------------------
// Kernel 2: G[t*32+b][n] = emb[inputs[b][t]] @ Wih.T + (bih+bhh)
// M = 8192 (m = t*32+b), N = 2048, K = 512.  One wave -> 16x64 tile.
// Software pipelined: fragments for k+1 are loaded before WMMAs for k.
// ---------------------------------------------------------------------------
__global__ __launch_bounds__(256) void k_input_gemm(const int* __restrict__ inputs,
                                                    const float* __restrict__ emb,
                                                    const __bf16* __restrict__ WihB,
                                                    const float* __restrict__ bias,
                                                    float* __restrict__ G) {
  const int lane = threadIdx.x & 31;
  const int wave = threadIdx.x >> 5;
  const int wgid = blockIdx.x * 8 + wave;     // 16384 waves total
  const int mtile = wgid >> 5;                // 0..511
  const int n0 = (wgid & 31) * 64;            // 0..2047 step 64
  const int lo = lane & 15, hi = lane >> 4;

  // A row owned by this lane (lanes 0-15 and 16-31 hold the same rows,
  // different K chunks, per the 16-bit A layout).
  const int m = mtile * 16 + lo;
  const int t = m >> 5, b = m & 31;
  const int erow = inputs[b * SLEN + t];
  const float* ea = emb + (size_t)erow * DDIM;

  auto loadA = [&](int kk) -> Frag {
    Frag a;
    const float4* p0 = (const float4*)(ea + kk + hi * 8);
    const float4* p1 = (const float4*)(ea + kk + 16 + hi * 8);
    float4 f0 = p0[0], f1 = p0[1], f2 = p1[0], f3 = p1[1];
    a.u[0] = cvt2(f0.x, f0.y); a.u[1] = cvt2(f0.z, f0.w);
    a.u[2] = cvt2(f1.x, f1.y); a.u[3] = cvt2(f1.z, f1.w);
    a.u[4] = cvt2(f2.x, f2.y); a.u[5] = cvt2(f2.z, f2.w);
    a.u[6] = cvt2(f3.x, f3.y); a.u[7] = cvt2(f3.z, f3.w);
    return a;
  };
  auto loadB = [&](int kk, Frag* bb) {
    const int cb = kk + hi * 8;               // B: table K order (8 + 8 at +16), matches A
#pragma unroll
    for (int s = 0; s < 4; ++s) {
      const __bf16* Br = WihB + (size_t)(n0 + s * 16 + lo) * DDIM + cb;
      bb[s].h[0] = *(const v8bf*)(Br); bb[s].h[1] = *(const v8bf*)(Br + 16);
    }
  };

  v8f acc0 = {}, acc1 = {}, acc2 = {}, acc3 = {};
  Frag aC = loadA(0);
  Frag bC[4]; loadB(0, bC);
  for (int kk = 0; kk < DDIM; kk += 32) {
    const int kn = (kk + 32 < DDIM) ? kk + 32 : 0;   // branchless tail (safe reload)
    Frag aN = loadA(kn);
    Frag bN[4]; loadB(kn, bN);
    acc0 = wmma_bf16(aC.v, bC[0].v, acc0);
    acc1 = wmma_bf16(aC.v, bC[1].v, acc1);
    acc2 = wmma_bf16(aC.v, bC[2].v, acc2);
    acc3 = wmma_bf16(aC.v, bC[3].v, acc3);
    aC = aN;
#pragma unroll
    for (int s = 0; s < 4; ++s) bC[s] = bN[s];
  }
  // Store D + bias.  C/D layout: n = n0+s*16+lo, m = mtile*16 + hi*8 + r.
#pragma unroll
  for (int s = 0; s < 4; ++s) {
    const int n = n0 + s * 16 + lo;
    const float bn = bias[n];
    v8f acc = (s == 0) ? acc0 : (s == 1) ? acc1 : (s == 2) ? acc2 : acc3;
#pragma unroll
    for (int r = 0; r < 8; ++r) {
      const int mr = mtile * 16 + hi * 8 + r;
      G[(size_t)mr * GDIM + n] = acc[r] + bn;
    }
  }
}

// ---------------------------------------------------------------------------
// Grid barrier: monotonic counter, zeroed by k_prep each launch.
// ---------------------------------------------------------------------------
__device__ __forceinline__ void grid_barrier(unsigned* cnt, volatile unsigned* gen,
                                             unsigned step) {
  __syncthreads();
  if (threadIdx.x == 0) {
    __threadfence();
    unsigned arrived = atomicAdd(cnt, 1u) + 1u;
    if (arrived == step * NWG) {
      __threadfence();
      *gen = step;
    } else {
      while (*gen < step) { __builtin_amdgcn_s_sleep(1); }
    }
  }
  __syncthreads();
  __threadfence();
}

// ---------------------------------------------------------------------------
// Kernel 3: persistent lattice recurrence.  32 WGs x 256 threads.
// WG g owns h columns [g*16, g*16+16); its Whh slice (64 gate cols x 512)
// is staged once into LDS and reused across all 256 steps.  Split-K gives
// each wave two independent WMMA accumulation chains.
// ---------------------------------------------------------------------------
__global__ __launch_bounds__(256) void k_recurrent(const int* __restrict__ prev_idx,
                                                   const float* __restrict__ prev_w,
                                                   const __bf16* __restrict__ WhhB,
                                                   const float* __restrict__ G,
                                                   float* __restrict__ hs,
                                                   float* __restrict__ cs,
                                                   unsigned* __restrict__ syncp) {
  __shared__ __attribute__((aligned(16))) __bf16 Bs[64][520];   // Whh slice (~65 KB)
  __shared__ __attribute__((aligned(16))) __bf16 hA[BDIM][520]; // mixed h, bf16
  __shared__ float cmix[BDIM][16];
  __shared__ float gbuf[4][BDIM][16];

  const int tid  = threadIdx.x;
  const int lane = tid & 31;
  const int wave = tid >> 5;
  const int lo = lane & 15, hi = lane >> 4;
  const int jbase = blockIdx.x * 16;
  unsigned* cnt = syncp;
  volatile unsigned* gen = syncp + 1;

  // ---- one-time: stage this WG's Whh slice into LDS (64 rows x 512 bf16) ----
  for (int e = tid; e < 64 * 256; e += 256) {
    const int lc = e >> 8, cu = e & 255;               // local row, dword col
    const int ngr = (lc >> 4) * HDIM + jbase + (lc & 15);
    *(unsigned*)&Bs[lc][cu * 2] =
        ((const unsigned*)(WhhB + (size_t)ngr * HDIM))[cu];
  }

  // per-wave GEMM tile: gate q (0..3), M-half mt (0..1); single 16x16 tile.
  const int q  = wave >> 1;
  const int mt = wave & 1;
  const int lc = q * 16 + lo;                 // LDS B row  (gate column, local)
  const int ng = q * HDIM + jbase + lo;       // global gate column
  const int arow = mt * 16 + lo;              // A row in hA

  auto loadB3 = [&](int kk) -> Frag {
    Frag bb; const int cb = kk + hi * 8;
    bb.h[0] = *(const v8bf*)&Bs[lc][cb]; bb.h[1] = *(const v8bf*)&Bs[lc][cb + 16];
    return bb;
  };
  auto loadA3 = [&](int kk) -> Frag {
    Frag a; const int c0 = kk + hi * 8;
    a.h[0] = *(const v8bf*)&hA[arow][c0]; a.h[1] = *(const v8bf*)&hA[arow][c0 + 16];
    return a;
  };

  for (int t = 0; t < SLEN; ++t) {
    // ---- phase 1: lattice-mix h (full, packed bf16 into LDS), c (own slice) ----
    if (t == 0) {
      for (int e = tid; e < BDIM * HDIM / 2; e += 256)
        *(unsigned*)&hA[e >> 8][(e & 255) * 2] = 0u;
      for (int e = tid; e < BDIM * 16; e += 256) cmix[e >> 4][e & 15] = 0.0f;
    } else {
      for (int e = tid; e < BDIM * HDIM / 2; e += 256) {
        const int b = e >> 8, j = (e & 255) * 2;
        const size_t base = ((size_t)b * SLEN + t) * KLAT;
        float a0 = 0.0f, a1 = 0.0f;
#pragma unroll
        for (int k = 0; k < KLAT; ++k) {
          const int idx = prev_idx[base + k];
          const float w = prev_w[base + k];
          const float* hp = hs + ((size_t)idx * BDIM + b) * HDIM + j;
          a0 += w * hp[0];
          a1 += w * hp[1];
        }
        *(unsigned*)&hA[b][j] = cvt2(a0, a1);
      }
      for (int e = tid; e < BDIM * 16; e += 256) {
        const int b = e >> 4, jj = e & 15;
        const size_t base = ((size_t)b * SLEN + t) * KLAT;
        float acc = 0.0f;
#pragma unroll
        for (int k = 0; k < KLAT; ++k) {
          const int idx = prev_idx[base + k];
          acc += prev_w[base + k] * cs[((size_t)idx * BDIM + b) * HDIM + jbase + jj];
        }
        cmix[b][jj] = acc;
      }
    }
    __syncthreads();

    // ---- phase 2: gates slice = hA @ Whh.T (+G); all operands from LDS.
    //      Split-K: two independent WMMA chains (K 0..255 and K 256..511). ----
    v8f accA = {}, accB = {};
    Frag bA = loadB3(0),   aA = loadA3(0);
    Frag bB = loadB3(256), aB = loadA3(256);
    for (int kk = 0; kk < HDIM / 2; kk += 32) {
      const int kn = (kk + 32 < HDIM / 2) ? kk + 32 : 0;
      Frag bAn = loadB3(kn),       aAn = loadA3(kn);
      Frag bBn = loadB3(kn + 256), aBn = loadA3(kn + 256);
      accA = wmma_bf16(aA.v, bA.v, accA);
      accB = wmma_bf16(aB.v, bB.v, accB);
      bA = bAn; aA = aAn; bB = bBn; aB = aBn;
    }
    {
      const float* Gt = G + (size_t)t * BDIM * GDIM;
      if (t + 1 < SLEN)
        __builtin_prefetch(G + ((size_t)(t + 1) * BDIM) * GDIM + ng, 0, 1);
#pragma unroll
      for (int r = 0; r < 8; ++r) {
        const int br = mt * 16 + hi * 8 + r;      // batch row 0..31
        gbuf[q][br][lo] = accA[r] + accB[r] + Gt[(size_t)br * GDIM + ng];
      }
    }
    __syncthreads();

    // ---- phase 3: LSTM cell nonlinearity for own 16 columns ----
    for (int e = tid; e < BDIM * 16; e += 256) {
      const int b = e >> 4, jj = e & 15;
      const float iv = gbuf[0][b][jj];
      const float fv = gbuf[1][b][jj];
      const float gv = gbuf[2][b][jj];
      const float ov = gbuf[3][b][jj];
      const float cn = sigm(fv) * cmix[b][jj] + sigm(iv) * tanhf(gv);
      const float hn = sigm(ov) * tanhf(cn);
      const size_t o = ((size_t)t * BDIM + b) * HDIM + jbase + jj;
      hs[o] = hn;
      cs[o] = cn;
    }
    grid_barrier(cnt, gen, (unsigned)(t + 1));
  }
}

// ---------------------------------------------------------------------------
// Kernel 4: masked temporal max-pool + final linear.  One block per batch.
// ---------------------------------------------------------------------------
__global__ __launch_bounds__(128) void k_pool_linear(const int* __restrict__ positions,
                                                     const float* __restrict__ hs,
                                                     const float* __restrict__ W_lin,
                                                     const float* __restrict__ b_lin,
                                                     float* __restrict__ out) {
  __shared__ float pooled[HDIM];
  const int b = blockIdx.x;
  const int tid = threadIdx.x;
  const int start = positions[b * 2];
  const int end   = positions[b * 2 + 1];
  for (int j = tid; j < HDIM; j += 128) {
    float m = -1e30f;
    for (int t = start; t < end; ++t)
      m = fmaxf(m, hs[((size_t)t * BDIM + b) * HDIM + j]);
    pooled[j] = (end > start) ? m : 0.0f;
  }
  __syncthreads();
  float acc = b_lin[tid];
  const float* wr = W_lin + (size_t)tid * HDIM;
  for (int j = 0; j < HDIM; ++j) acc += pooled[j] * wr[j];
  out[b * LOUT + tid] = acc;
}

// ---------------------------------------------------------------------------
extern "C" void kernel_launch(void* const* d_in, const int* in_sizes, int n_in,
                              void* d_out, int out_size, void* d_ws, size_t ws_size,
                              hipStream_t stream) {
  const int*   inputs    = (const int*)d_in[0];
  const int*   positions = (const int*)d_in[1];
  const int*   prev_idx  = (const int*)d_in[2];
  const float* prev_w    = (const float*)d_in[3];
  const float* emb       = (const float*)d_in[4];
  const float* Wih       = (const float*)d_in[5];
  const float* Whh       = (const float*)d_in[6];
  const float* bih       = (const float*)d_in[7];
  const float* bhh       = (const float*)d_in[8];
  const float* W_lin     = (const float*)d_in[9];
  const float* b_lin     = (const float*)d_in[10];
  float* out = (float*)d_out;

  // workspace layout (~100 MB, all L2-resident)
  unsigned* syncp = (unsigned*)d_ws;
  __bf16* WihB = (__bf16*)((char*)d_ws + 256);
  __bf16* WhhB = WihB + (size_t)GDIM * DDIM;
  float*  bias = (float*)(WhhB + (size_t)GDIM * HDIM);
  float*  G    = bias + GDIM;
  float*  hs   = G + (size_t)SLEN * BDIM * GDIM;
  float*  cs   = hs + (size_t)SLEN * BDIM * HDIM;

  k_prep<<<512, 256, 0, stream>>>(Wih, Whh, bih, bhh,
                                  (unsigned*)WihB, (unsigned*)WhhB, bias, syncp);
  k_input_gemm<<<2048, 256, 0, stream>>>(inputs, emb, WihB, bias, G);
  k_recurrent<<<NWG, 256, 0, stream>>>(prev_idx, prev_w, WhhB, G, hs, cs, syncp);
  k_pool_linear<<<BDIM, 128, 0, stream>>>(positions, hs, W_lin, b_lin, out);
}
